// language_8057358647401
// MI455X (gfx1250) — matrix operand order repaired and checked by reading, not compiled, not run on hardware
//
#include <hip/hip_runtime.h>
#include <hip/hip_bf16.h>
#include <math.h>

typedef __attribute__((ext_vector_type(16))) _Float16 v16h;
typedef __attribute__((ext_vector_type(8)))  _Float16 v8h;
typedef __attribute__((ext_vector_type(8)))  float    v8f;

#define Dd   1024
#define Bb   32
#define Tt   64
#define G4   4096
#define LSTM_BLOCKS 64
#define LSTM_THREADS 256

// ---------------- helpers ----------------

// A-matrix (16x32 f16, M x K) fragment load, contiguous k order:
// lanes 0-15 (m=lane): halves 0..15 -> K=0..15
// lanes 16-31 (m=lane-16): halves 0..15 -> K=16..31
// => two contiguous 8-half (16B) chunks at +ak and +ak+8, ak = (lane>=16)?16:0
__device__ __forceinline__ v16h load_a16(const _Float16* p, int ak) {
  union { v16h v; v8h h[2]; } u;
  u.h[0] = *(const v8h*)(p + ak);
  u.h[1] = *(const v8h*)(p + ak + 8);
  return u.v;
}

#define WMMA_F16(a, b, c) \
  __builtin_amdgcn_wmma_f32_16x16x32_f16(false, (a), false, (b), (short)0, (c), false, false)

// Async global->LDS copy of one 16B chunk per lane (CDNA5 async path, ASYNCcnt).
// LDS dest address = low 32 bits of generic LDS pointer (workgroup-relative offset).
__device__ __forceinline__ void async_copy16(unsigned lds_off, const void* gptr) {
  asm volatile("global_load_async_to_lds_b128 %0, %1, off"
               :: "v"(lds_off), "v"((unsigned long long)(uintptr_t)gptr)
               : "memory");
}
__device__ __forceinline__ void wait_async0() {
  asm volatile("s_wait_asynccnt 0x0" ::: "memory");
}

__device__ __forceinline__ float sigf(float x) { return 1.0f / (1.0f + __expf(-x)); }

__device__ __forceinline__ void grid_barrier(unsigned* cnt, unsigned* gen, unsigned nblocks) {
  __threadfence();
  __syncthreads();
  if (threadIdx.x == 0) {
    unsigned g = __hip_atomic_load(gen, __ATOMIC_ACQUIRE, __HIP_MEMORY_SCOPE_AGENT);
    unsigned a = __hip_atomic_fetch_add(cnt, 1u, __ATOMIC_ACQ_REL, __HIP_MEMORY_SCOPE_AGENT);
    if (a == nblocks - 1) {
      __hip_atomic_store(cnt, 0u, __ATOMIC_RELAXED, __HIP_MEMORY_SCOPE_AGENT);
      __hip_atomic_fetch_add(gen, 1u, __ATOMIC_ACQ_REL, __HIP_MEMORY_SCOPE_AGENT);
    } else {
      while (__hip_atomic_load(gen, __ATOMIC_ACQUIRE, __HIP_MEMORY_SCOPE_AGENT) == g) { }
    }
  }
  __syncthreads();
}

// ---------------- kernels ----------------

__global__ void k_f32_to_f16(const float* __restrict__ in, _Float16* __restrict__ out, int n) {
  int i = blockIdx.x * blockDim.x + threadIdx.x;
  if (i < n) out[i] = (_Float16)in[i];
}

__global__ void k_init_sync(unsigned* sync) {
  if (threadIdx.x == 0) { sync[0] = 0u; sync[1] = 0u; }
}

// xg[t*Bb+b][g] = Q[b][t][:] . W_ih[g][:] + b_ih[g] + b_hh[g]
// M=2048 (row r: t=r>>5, b=r&31), N=4096, K=1024.
// Block = 8 waves sharing one 16-row A panel (staged async to LDS);
// each wave computes a 16x64 C strip (4 N-tiles, A fragment reused 4x).
// Grid: 128 tm * 8 = 1024 blocks.
__global__ void k_gemm_xgates(const _Float16* __restrict__ Qh,
                              const _Float16* __restrict__ Wih,
                              const float* __restrict__ bih,
                              const float* __restrict__ bhh,
                              float* __restrict__ xg) {
  __shared__ _Float16 lds_a[16 * Dd];          // 32 KB A panel
  const int tid  = threadIdx.x;
  const int lane = tid & 31;
  const int warp = tid >> 5;
  const int tm   = blockIdx.x >> 3;            // 0..127
  const int ng   = (blockIdx.x & 7) * 8 + warp;// 0..63, each = 64 output cols
  const int m    = lane & 15;
  const bool hi  = lane >= 16;
  const int ak   = hi ? 16 : 0;
  const int bk   = hi ? 16 : 0;

  // async stage A panel: 16 rows x 1024 halves = 2048 x 16B chunks
  const unsigned lbase = (unsigned)(uintptr_t)lds_a;
  for (int c = tid; c < 2048; c += 256) {
    const int row = c >> 7, col = c & 127;
    const int r = tm * 16 + row;
    const int t = r >> 5, b = r & 31;
    async_copy16(lbase + (unsigned)c * 16u,
                 Qh + ((size_t)(b * Tt + t)) * Dd + col * 8);
  }
  wait_async0();
  __syncthreads();

  const _Float16* arow  = lds_a + (size_t)m * Dd;
  const _Float16* brow0 = Wih + ((size_t)(ng * 64 + m)) * Dd;   // col n = lane&15 of B^T

  v8f acc0 = {}, acc1 = {}, acc2 = {}, acc3 = {};
  for (int k0 = 0; k0 < Dd; k0 += 32) {
    v16h a  = load_a16(arow + k0, ak);                          // ds_load_b128 x2
    v16h b0 = *(const v16h*)(brow0 + (size_t)0  * Dd + k0 + bk);
    v16h b1 = *(const v16h*)(brow0 + (size_t)16 * Dd + k0 + bk);
    v16h b2 = *(const v16h*)(brow0 + (size_t)32 * Dd + k0 + bk);
    v16h b3 = *(const v16h*)(brow0 + (size_t)48 * Dd + k0 + bk);
    acc0 = WMMA_F16(a, b0, acc0);
    acc1 = WMMA_F16(a, b1, acc1);
    acc2 = WMMA_F16(a, b2, acc2);
    acc3 = WMMA_F16(a, b3, acc3);
  }

  const int rb = tm * 16 + (hi ? 8 : 0);
  v8f accs[4] = {acc0, acc1, acc2, acc3};
#pragma unroll
  for (int j = 0; j < 4; ++j) {
    const int nc = ng * 64 + j * 16 + (lane & 15);
    const float bia = bih[nc] + bhh[nc];
#pragma unroll
    for (int v = 0; v < 8; ++v)
      xg[(size_t)(rb + v) * G4 + nc] = accs[j][v] + bia;
  }
}

// Persistent LSTM scan: 64 blocks x 256 threads (512 waves == 512 gate tiles).
// Each block async-stages its 16-row h panel to LDS every step.
__global__ void k_lstm_scan(_Float16* __restrict__ attin16,      // [Bb*Tt, Dd] f16 h states
                            const _Float16* __restrict__ Whh,    // [G4, Dd] f16
                            const float* __restrict__ xg,        // [Tt, Bb, G4]
                            float* __restrict__ gate,            // [Bb, G4] scratch
                            float* __restrict__ cstate,          // [Bb, Dd]
                            float* __restrict__ attin_out,       // [Bb, Tt, Dd] f32 (d_out)
                            unsigned* __restrict__ sync) {
  __shared__ _Float16 lds_a[16 * Dd];          // 32 KB h panel
  const int tid  = threadIdx.x;
  const int lane = tid & 31;
  const int warp = tid >> 5;
  const int wid  = blockIdx.x * 8 + warp;      // 0..511
  const int tm   = wid >> 8;                   // 0..1 (uniform within a block)
  const int tn   = wid & 255;                  // 0..255
  const bool hi  = lane >= 16;
  const int ak   = hi ? 16 : 0;
  const int bk   = hi ? 16 : 0;
  const int m    = lane & 15;
  const _Float16* brow = Whh + (size_t)(tn * 16 + m) * Dd;
  const _Float16* arow = lds_a + (size_t)m * Dd;
  const int nc = tn * 16 + (lane & 15);
  const int rb = tm * 16 + (hi ? 8 : 0);
  const unsigned lbase = (unsigned)(uintptr_t)lds_a;
  unsigned* cnt = sync;
  unsigned* gen = sync + 1;
  const int gidx = blockIdx.x * blockDim.x + tid;   // 0..16383

  for (int t = 0; t < Tt; ++t) {
    if (t > 0) {
      // async stage h(t-1) rows [tm*16, tm*16+16) into LDS
      for (int c = tid; c < 2048; c += 256) {
        const int row = c >> 7, col = c & 127;
        async_copy16(lbase + (unsigned)c * 16u,
                     attin16 + ((size_t)((tm * 16 + row) * Tt + (t - 1))) * Dd + col * 8);
      }
      wait_async0();
      __syncthreads();

      v8f acc = {};
#pragma unroll 4
      for (int k0 = 0; k0 < Dd; k0 += 32) {
        v16h a  = load_a16(arow + k0, ak);                 // from LDS
        v16h bm = *(const v16h*)(brow + k0 + bk);          // W_hh from L2
        acc = WMMA_F16(a, bm, acc);
      }
      const float* xgt = xg + (size_t)t * Bb * G4;
#pragma unroll
      for (int v = 0; v < 8; ++v)
        gate[(size_t)(rb + v) * G4 + nc] = acc[v] + xgt[(size_t)(rb + v) * G4 + nc];
    }
    grid_barrier(cnt, gen, LSTM_BLOCKS);

    // elementwise gates: 32*1024 elements over 16384 threads (2 each)
    const float* pre = (t == 0) ? xg : gate;   // xg[t=0] block has identical [Bb,G4] layout
    for (int e = gidx; e < Bb * Dd; e += LSTM_BLOCKS * LSTM_THREADS) {
      const int b = e >> 10, d = e & 1023;
      const size_t base = (size_t)b * G4;
      const float iv = sigf(pre[base + d]);
      const float fv = sigf(pre[base + Dd + d]);
      const float gv = tanhf(pre[base + 2 * Dd + d]);
      const float ov = sigf(pre[base + 3 * Dd + d]);
      const float c  = (t == 0) ? (iv * gv) : (fv * cstate[e] + iv * gv);
      cstate[e] = c;
      const float h = ov * tanhf(c);
      const size_t oi = ((size_t)b * Tt + t) * Dd + d;
      attin_out[oi] = h;
      attin16[oi]   = (_Float16)h;
    }
    grid_barrier(cnt, gen, LSTM_BLOCKS);
  }
}

// att_w = tanh(att_in @ W_att^T + b_att); M=2048, N=1024, K=1024.
// Same blocking as k_gemm_xgates: LDS-staged A panel, 16x64 strip per wave.
// Grid: 128 tm * 2 = 256 blocks.
__global__ void k_gemm_attw(const _Float16* __restrict__ Ah,     // attin16 [2048, Dd]
                            const _Float16* __restrict__ Watt,   // [Dd, Dd]
                            const float* __restrict__ batt,
                            float* __restrict__ attw) {
  __shared__ _Float16 lds_a[16 * Dd];
  const int tid  = threadIdx.x;
  const int lane = tid & 31;
  const int warp = tid >> 5;
  const int tm   = blockIdx.x >> 1;                 // 0..127
  const int ng   = (blockIdx.x & 1) * 8 + warp;     // 0..15, each = 64 cols
  const int m    = lane & 15;
  const bool hi  = lane >= 16;
  const int ak   = hi ? 16 : 0;
  const int bk   = hi ? 16 : 0;

  const unsigned lbase = (unsigned)(uintptr_t)lds_a;
  for (int c = tid; c < 2048; c += 256) {
    const int row = c >> 7, col = c & 127;
    async_copy16(lbase + (unsigned)c * 16u,
                 Ah + ((size_t)(tm * 16 + row)) * Dd + col * 8);
  }
  wait_async0();
  __syncthreads();

  const _Float16* arow  = lds_a + (size_t)m * Dd;
  const _Float16* brow0 = Watt + ((size_t)(ng * 64 + m)) * Dd;

  v8f acc0 = {}, acc1 = {}, acc2 = {}, acc3 = {};
  for (int k0 = 0; k0 < Dd; k0 += 32) {
    v16h a  = load_a16(arow + k0, ak);
    v16h b0 = *(const v16h*)(brow0 + (size_t)0  * Dd + k0 + bk);
    v16h b1 = *(const v16h*)(brow0 + (size_t)16 * Dd + k0 + bk);
    v16h b2 = *(const v16h*)(brow0 + (size_t)32 * Dd + k0 + bk);
    v16h b3 = *(const v16h*)(brow0 + (size_t)48 * Dd + k0 + bk);
    acc0 = WMMA_F16(a, b0, acc0);
    acc1 = WMMA_F16(a, b1, acc1);
    acc2 = WMMA_F16(a, b2, acc2);
    acc3 = WMMA_F16(a, b3, acc3);
  }

  const int rb = tm * 16 + (hi ? 8 : 0);
  v8f accs[4] = {acc0, acc1, acc2, acc3};
#pragma unroll
  for (int j = 0; j < 4; ++j) {
    const int nc = ng * 64 + j * 16 + (lane & 15);
    const float bia = batt[nc];
#pragma unroll
    for (int v = 0; v < 8; ++v)
      attw[(size_t)(rb + v) * Dd + nc] = tanhf(accs[j][v] + bia);
  }
}

// scores[r] = attw[r,:] . W_ctx + b_ctx   (one block per row r = b*Tt+t)
__global__ void k_scores(const float* __restrict__ attw, const float* __restrict__ Wctx,
                         const float* __restrict__ bctx, float* __restrict__ scores) {
  __shared__ float red[256];
  const int r = blockIdx.x;
  float s = 0.f;
  for (int d = threadIdx.x; d < Dd; d += 256)
    s += attw[(size_t)r * Dd + d] * Wctx[d];
  red[threadIdx.x] = s;
  __syncthreads();
  for (int off = 128; off > 0; off >>= 1) {
    if (threadIdx.x < off) red[threadIdx.x] += red[threadIdx.x + off];
    __syncthreads();
  }
  if (threadIdx.x == 0) scores[r] = red[0] + bctx[0];
}

// softmax over T per batch + weighted pooling of att_in -> att_out (f32 + f16)
__global__ void k_attn_pool(const float* __restrict__ scores, const float* __restrict__ attin,
                            float* __restrict__ attout, _Float16* __restrict__ attout16) {
  __shared__ float w[Tt];
  const int b = blockIdx.x;
  if (threadIdx.x == 0) {
    float mx = -1e30f;
    for (int t = 0; t < Tt; ++t) mx = fmaxf(mx, scores[b * Tt + t]);
    float sum = 0.f;
    for (int t = 0; t < Tt; ++t) { float e = __expf(scores[b * Tt + t] - mx); w[t] = e; sum += e; }
    const float inv = 1.f / sum;
    for (int t = 0; t < Tt; ++t) w[t] *= inv;
  }
  __syncthreads();
  for (int d = threadIdx.x; d < Dd; d += 256) {
    float s = 0.f;
    for (int t = 0; t < Tt; ++t)
      s += w[t] * attin[((size_t)b * Tt + t) * Dd + d];
    attout[(size_t)b * Dd + d]   = s;
    attout16[(size_t)b * Dd + d] = (_Float16)s;
  }
}

// final = relu(att_out @ W_lang^T + b_lang); M=32, N=1024, K=1024
__global__ void k_gemm_final(const _Float16* __restrict__ Ah, const _Float16* __restrict__ Wl,
                             const float* __restrict__ bl, float* __restrict__ out) {
  const int lane = threadIdx.x & 31;
  const int warp = threadIdx.x >> 5;
  const int wid  = blockIdx.x * 8 + warp;     // 0..127
  const int tm   = wid >> 6;                  // 0..1
  const int tn   = wid & 63;                  // 0..63
  const int m    = lane & 15;
  const bool hi  = lane >= 16;
  const int ak   = hi ? 16 : 0;
  const int bk   = hi ? 16 : 0;

  const _Float16* arow = Ah + (size_t)(tm * 16 + m) * Dd;
  const _Float16* brow = Wl + (size_t)(tn * 16 + m) * Dd;

  v8f acc = {};
  for (int k0 = 0; k0 < Dd; k0 += 32) {
    v16h a  = load_a16(arow + k0, ak);
    v16h bm = *(const v16h*)(brow + k0 + bk);
    acc = WMMA_F16(a, bm, acc);
  }
  const int nc = tn * 16 + (lane & 15);
  const int rb = tm * 16 + (hi ? 8 : 0);
  const float bia = bl[nc];
#pragma unroll
  for (int v = 0; v < 8; ++v)
    out[(size_t)(rb + v) * Dd + nc] = fmaxf(acc[v] + bia, 0.f);
}

// ---------------- launch ----------------

extern "C" void kernel_launch(void* const* d_in, const int* in_sizes, int n_in,
                              void* d_out, int out_size, void* d_ws, size_t ws_size,
                              hipStream_t stream) {
  const float* Q     = (const float*)d_in[0];
  const float* W_ih  = (const float*)d_in[1];
  const float* W_hh  = (const float*)d_in[2];
  const float* b_ih  = (const float*)d_in[3];
  const float* b_hh  = (const float*)d_in[4];
  const float* W_att = (const float*)d_in[5];
  const float* b_att = (const float*)d_in[6];
  const float* W_ctx = (const float*)d_in[7];
  const float* b_ctx = (const float*)d_in[8];
  const float* W_lang= (const float*)d_in[9];
  const float* b_lang= (const float*)d_in[10];

  float* out_final = (float*)d_out;                 // [32, 1024]
  float* out_attin = (float*)d_out + Bb * Dd;       // [32, 64, 1024]

  // workspace carve-up (all region sizes 256B-aligned)
  char* ws = (char*)d_ws;
  size_t off = 0;
  unsigned*  sync     = (unsigned*)(ws + off);  off += 256;
  _Float16*  Wih16    = (_Float16*)(ws + off);  off += (size_t)G4 * Dd * 2;     // 8 MB
  _Float16*  Whh16    = (_Float16*)(ws + off);  off += (size_t)G4 * Dd * 2;     // 8 MB
  _Float16*  Watt16   = (_Float16*)(ws + off);  off += (size_t)Dd * Dd * 2;     // 2 MB
  _Float16*  Wlang16  = (_Float16*)(ws + off);  off += (size_t)Dd * Dd * 2;     // 2 MB
  _Float16*  Q16      = (_Float16*)(ws + off);  off += (size_t)Bb * Tt * Dd * 2;// 4 MB
  _Float16*  attin16  = (_Float16*)(ws + off);  off += (size_t)Bb * Tt * Dd * 2;// 4 MB
  float*     xg       = (float*)(ws + off);     off += (size_t)Tt * Bb * G4 * 4;// 32 MB
  float*     gate     = (float*)(ws + off);     off += (size_t)Bb * G4 * 4;     // 512 KB
  float*     cstate   = (float*)(ws + off);     off += (size_t)Bb * Dd * 4;     // 128 KB
  float*     attw     = (float*)(ws + off);     off += (size_t)Bb * Tt * Dd * 4;// 8 MB
  float*     scores   = (float*)(ws + off);     off += (size_t)Bb * Tt * 4 + 256;
  float*     attout   = (float*)(ws + off);     off += (size_t)Bb * Dd * 4;     // 128 KB
  _Float16*  attout16 = (_Float16*)(ws + off);  off += (size_t)Bb * Dd * 2;     // 64 KB
  (void)ws_size; (void)out_size; (void)n_in; (void)in_sizes;

  // 1) f32 -> f16 conversions
  {
    int n;
    n = G4 * Dd;      k_f32_to_f16<<<(n + 255) / 256, 256, 0, stream>>>(W_ih,  Wih16,   n);
    n = G4 * Dd;      k_f32_to_f16<<<(n + 255) / 256, 256, 0, stream>>>(W_hh,  Whh16,   n);
    n = Dd * Dd;      k_f32_to_f16<<<(n + 255) / 256, 256, 0, stream>>>(W_att, Watt16,  n);
    n = Dd * Dd;      k_f32_to_f16<<<(n + 255) / 256, 256, 0, stream>>>(W_lang,Wlang16, n);
    n = Bb * Tt * Dd; k_f32_to_f16<<<(n + 255) / 256, 256, 0, stream>>>(Q,     Q16,     n);
  }

  // 2) x_gates GEMM: 128 tm-panels x 8 blocks, 16x64 strip per wave
  k_gemm_xgates<<<1024, 256, 0, stream>>>(Q16, Wih16, b_ih, b_hh, xg);

  // 3) persistent LSTM scan with grid barrier + async LDS staging of h
  k_init_sync<<<1, 32, 0, stream>>>(sync);
  k_lstm_scan<<<LSTM_BLOCKS, LSTM_THREADS, 0, stream>>>(attin16, Whh16, xg, gate, cstate,
                                                        out_attin, sync);

  // 4) attention tanh GEMM: 128 tm-panels x 2 blocks
  k_gemm_attw<<<256, 256, 0, stream>>>(attin16, Watt16, b_att, attw);

  // 5) context scores + softmax pooling
  k_scores<<<Bb * Tt, 256, 0, stream>>>(attw, W_ctx, b_ctx, scores);
  k_attn_pool<<<Bb, 256, 0, stream>>>(scores, out_attin, attout, attout16);

  // 6) language head GEMM + ReLU: 128 tiles
  k_gemm_final<<<16, 256, 0, stream>>>(attout16, Wlang16, b_lang, out_final);
}
